// BahdanauAttention_66494683676890
// MI455X (gfx1250) — matrix operand order repaired and checked by reading, not compiled, not run on hardware
//
#include <hip/hip_runtime.h>
#include <hip/hip_bf16.h>
#include <stdint.h>

typedef __attribute__((ext_vector_type(16))) __bf16 v16bf;
typedef __attribute__((ext_vector_type(8)))  float  v8f;
typedef __attribute__((ext_vector_type(8)))  unsigned int v8u;
typedef __attribute__((ext_vector_type(4)))  float  v4f;

#define B_  32
#define L_  2048
#define H_  1024
#define F_  1024

// Hardware tanh (CDNA5 V_TANH_F32 transcendental) when the builtin exists.
static __device__ __forceinline__ float fast_tanh(float x) {
#if __has_builtin(__builtin_amdgcn_tanhf)
    return __builtin_amdgcn_tanhf(x);
#elif __has_builtin(__builtin_amdgcn_tanh_f32)
    return __builtin_amdgcn_tanh_f32(x);
#else
    return tanhf(x);
#endif
}

static __device__ __forceinline__ unsigned short f32_to_bf16(float x) {
    unsigned int u = __builtin_bit_cast(unsigned int, x);
    unsigned int r = u + 0x7FFFu + ((u >> 16) & 1u);   // round-to-nearest-even
    return (unsigned short)(r >> 16);
}

// ---------------------------------------------------------------------------
// Kernel 1: h_proj[b,h] = sum_k hidden[b,k] * W_h[k,h]   (tiny, VALU)
// ---------------------------------------------------------------------------
__global__ __launch_bounds__(256) void hproj_kernel(
    const float* __restrict__ hidden, const float* __restrict__ Wh,
    float* __restrict__ hproj)
{
    int b  = blockIdx.x;      // 32
    int h0 = threadIdx.x;     // 256 threads, 4 h each (coalesced over h)
    float a0 = 0.f, a1 = 0.f, a2 = 0.f, a3 = 0.f;
    const float* hrow = hidden + b * H_;
    for (int k = 0; k < H_; ++k) {
        float hv = hrow[k];
        const float* wr = Wh + (size_t)k * H_;
        a0 += hv * wr[h0];
        a1 += hv * wr[h0 + 256];
        a2 += hv * wr[h0 + 512];
        a3 += hv * wr[h0 + 768];
    }
    float* o = hproj + b * H_;
    o[h0] = a0; o[h0 + 256] = a1; o[h0 + 512] = a2; o[h0 + 768] = a3;
}

// ---------------------------------------------------------------------------
// Kernel 2: pack W_f (f32, row-major [F,H]) into bf16 WMMA B-fragment order.
// Fragment (nt, ks): 32 lanes x 16 bf16.  B layout (ISA 7.12.2, 16-bit B 32x16):
//   lane l: n = nt*16 + (l & 15),  k = ks*32 + (l>>4)*8 + (j&7) + (j>>3)*16,  j = 0..15
// ---------------------------------------------------------------------------
__global__ __launch_bounds__(256) void packWf_kernel(
    const float* __restrict__ Wf, unsigned short* __restrict__ wp)
{
    int t    = blockIdx.x * 256 + threadIdx.x;   // 0 .. F*H-1
    int j    = t & 15;
    int lane = (t >> 4) & 31;
    int ks   = (t >> 9) & 31;
    int nt   = t >> 14;
    int k = ks * 32 + ((lane >> 4) * 8) + (j & 7) + ((j >> 3) * 16);
    int n = nt * 16 + (lane & 15);
    wp[t] = f32_to_bf16(Wf[(size_t)k * H_ + n]);
}

// ---------------------------------------------------------------------------
// Kernel 3: fused GEMM + tanh + V-dot -> energy[b*L + l]
// One wave = 16 rows of features. Stage rows in LDS as bf16 (32 KB/wave),
// then 64 N-tiles x 32 K-steps of v_wmma_f32_16x16x32_bf16.
// ---------------------------------------------------------------------------
__global__ __launch_bounds__(64) void energy_kernel(
    const float* __restrict__ features,
    const float* __restrict__ hproj,
    const float* __restrict__ Vv,
    const unsigned short* __restrict__ wp,
    float* __restrict__ energy)
{
    __shared__ unsigned short sA[2][16][H_];    // 2 waves * 16 rows * 1024 bf16 = 64 KB
    const int lane = threadIdx.x & 31;
    const int wv   = threadIdx.x >> 5;
    const int tile = blockIdx.x * 2 + wv;       // 0..4095 (16-row tiles)
    const int row0 = tile * 16;
    const int bidx = row0 / L_;                 // L_ % 16 == 0 -> one batch per tile

    // --- stage: 16 rows x 1024 f32 -> bf16 in LDS; streaming, non-temporal
    {
        const int m    = lane & 15;
        const int half = lane >> 4;             // each half-wave loads 512 cols
        const v4f* src = (const v4f*)(features + (size_t)(row0 + m) * F_ + half * 512);
        unsigned short* dst = &sA[wv][m][half * 512];
        for (int c = 0; c < 512; c += 4) {
            v4f v = __builtin_nontemporal_load(src + (c >> 2));
            dst[c + 0] = f32_to_bf16(v[0]);
            dst[c + 1] = f32_to_bf16(v[1]);
            dst[c + 2] = f32_to_bf16(v[2]);
            dst[c + 3] = f32_to_bf16(v[3]);
        }
    }
    __syncthreads();

    // A layout (16-bit A 16x32): lane<16 -> K {0..7},{16..23}; lane>=16 -> {8..15},{24..31}
    const int m     = lane & 15;
    const int khalf = (lane >> 4) * 8;
    float e[8] = {0.f, 0.f, 0.f, 0.f, 0.f, 0.f, 0.f, 0.f};

    for (int nt = 0; nt < 64; ++nt) {
        v8f acc = {0.f, 0.f, 0.f, 0.f, 0.f, 0.f, 0.f, 0.f};
        const unsigned short* bbase = wp + ((size_t)nt * 32 * 32 + lane) * 16;
        for (int ks = 0; ks < 32; ++ks) {
            const unsigned short* ap = &sA[wv][m][ks * 32 + khalf];
            uint4 a0 = *(const uint4*)ap;              // K .. K+7
            uint4 a1 = *(const uint4*)(ap + 16);       // K+16 .. K+23
            const unsigned short* bp = bbase + (size_t)ks * 32 * 16;
            uint4 b0 = *(const uint4*)bp;              // j = 0..7
            uint4 b1 = *(const uint4*)(bp + 8);        // j = 8..15
            v8u au, bu;
            au[0] = a0.x; au[1] = a0.y; au[2] = a0.z; au[3] = a0.w;
            au[4] = a1.x; au[5] = a1.y; au[6] = a1.z; au[7] = a1.w;
            bu[0] = b0.x; bu[1] = b0.y; bu[2] = b0.z; bu[3] = b0.w;
            bu[4] = b1.x; bu[5] = b1.y; bu[6] = b1.z; bu[7] = b1.w;
            v16bf av = __builtin_bit_cast(v16bf, au);
            v16bf bv = __builtin_bit_cast(v16bf, bu);
            acc = __builtin_amdgcn_wmma_f32_16x16x32_bf16(
                false, av, false, bv, (short)0, acc, false, false);
        }
        // epilogue: each lane owns column n of the C tile
        const int n = nt * 16 + (lane & 15);
        const float hp = hproj[bidx * H_ + n];
        const float vvn = Vv[n];
        #pragma unroll
        for (int i = 0; i < 8; ++i)
            e[i] += fast_tanh(acc[i] + hp) * vvn;
    }

    // reduce over the 16 columns of each half-wave (rows 0-7 in lanes 0-15,
    // rows 8-15 in lanes 16-31)
    #pragma unroll
    for (int i = 0; i < 8; ++i) {
        float s = e[i];
        s += __shfl_xor(s, 1, 16);
        s += __shfl_xor(s, 2, 16);
        s += __shfl_xor(s, 4, 16);
        s += __shfl_xor(s, 8, 16);
        e[i] = s;
    }
    if (lane == 0) {
        #pragma unroll
        for (int i = 0; i < 8; ++i) energy[row0 + i] = e[i];
    } else if (lane == 16) {
        #pragma unroll
        for (int i = 0; i < 8; ++i) energy[row0 + 8 + i] = e[i];
    }
}

// ---------------------------------------------------------------------------
// Kernel 4: softmax over L per batch
// ---------------------------------------------------------------------------
__global__ __launch_bounds__(256) void softmax_kernel(
    const float* __restrict__ energy, float* __restrict__ weights)
{
    __shared__ float red[256];
    const int b = blockIdx.x;
    const int t = threadIdx.x;
    const float* er = energy + (size_t)b * L_;

    float mx = -3.0e38f;
    for (int l = t; l < L_; l += 256) mx = fmaxf(mx, er[l]);
    red[t] = mx; __syncthreads();
    for (int off = 128; off > 0; off >>= 1) {
        if (t < off) red[t] = fmaxf(red[t], red[t + off]);
        __syncthreads();
    }
    mx = red[0]; __syncthreads();

    float sum = 0.f;
    for (int l = t; l < L_; l += 256) sum += __expf(er[l] - mx);
    red[t] = sum; __syncthreads();
    for (int off = 128; off > 0; off >>= 1) {
        if (t < off) red[t] += red[t + off];
        __syncthreads();
    }
    const float inv = 1.0f / red[0];
    for (int l = t; l < L_; l += 256)
        weights[(size_t)b * L_ + l] = __expf(er[l] - mx) * inv;
}

// ---------------------------------------------------------------------------
// Kernel 5: context[b,f] = sum_l weights[b,l] * features[b,l,f]
// weights broadcast from LDS; features streamed non-temporal (2nd HBM pass)
// ---------------------------------------------------------------------------
__global__ __launch_bounds__(256) void context_kernel(
    const float* __restrict__ features,
    const float* __restrict__ weights,
    float* __restrict__ context)
{
    __shared__ float w[L_];                 // 8 KB
    const int b = blockIdx.y;
    const int f = blockIdx.x * 256 + threadIdx.x;
    for (int l = threadIdx.x; l < L_; l += 256) w[l] = weights[(size_t)b * L_ + l];
    __syncthreads();
    float acc = 0.f;
    const float* fb = features + (size_t)b * L_ * F_ + f;
    for (int l = 0; l < L_; ++l)
        acc += w[l] * __builtin_nontemporal_load(fb + (size_t)l * F_);
    context[(size_t)b * F_ + f] = acc;
}

// ---------------------------------------------------------------------------
extern "C" void kernel_launch(void* const* d_in, const int* in_sizes, int n_in,
                              void* d_out, int out_size, void* d_ws, size_t ws_size,
                              hipStream_t stream)
{
    (void)in_sizes; (void)n_in; (void)out_size; (void)ws_size;
    const float* hidden   = (const float*)d_in[0];   // [B,H]
    const float* features = (const float*)d_in[1];   // [B,L,F]
    const float* Wh       = (const float*)d_in[2];   // [H,H]
    const float* Wf       = (const float*)d_in[3];   // [F,H]
    const float* V        = (const float*)d_in[4];   // [H]

    float* out = (float*)d_out;
    float* context_out = out;             // [B,F]  = 32768 floats
    float* weights_out = out + B_ * F_;   // [B,L]  = 65536 floats

    char* ws = (char*)d_ws;
    float* hproj            = (float*)ws;                                  // 128 KB
    float* energy           = (float*)(ws + 128 * 1024);                   // 256 KB
    unsigned short* wfpack  = (unsigned short*)(ws + (128 + 256) * 1024);  // 2 MB

    hipLaunchKernelGGL(hproj_kernel,   dim3(B_),              dim3(256), 0, stream,
                       hidden, Wh, hproj);
    hipLaunchKernelGGL(packWf_kernel,  dim3((F_ * H_) / 256), dim3(256), 0, stream,
                       Wf, wfpack);
    hipLaunchKernelGGL(energy_kernel,  dim3((B_ * L_) / 32),  dim3(64), 0, stream,
                       features, hproj, V, wfpack, energy);
    hipLaunchKernelGGL(softmax_kernel, dim3(B_),              dim3(256), 0, stream,
                       energy, weights_out);
    hipLaunchKernelGGL(context_kernel, dim3(F_ / 256, B_),    dim3(256), 0, stream,
                       features, weights_out, context_out);
}
